// CustomT5ForSummarization_83296595738970
// MI455X (gfx1250) — matrix operand order repaired and checked by reading, not compiled, not run on hardware
//
#include <hip/hip_runtime.h>
#include <hip/hip_bf16.h>
#include <math.h>
#include <stdint.h>

// CDNA5 async global->LDS copies (GLOBAL_LOAD_ASYNC_TO_LDS_B128, ASYNCcnt).
// Set to 0 to fall back to the proven global->VGPR->ds_store staging path.
#define USE_ASYNC_LDS 1

typedef _Float16 half_t;
typedef _Float16 v16h __attribute__((ext_vector_type(16)));
typedef _Float16 v8h  __attribute__((ext_vector_type(8)));
typedef float    v8f  __attribute__((ext_vector_type(8)));

namespace {
constexpr int kV = 32000, kD = 512, kH = 8, kL = 6, kDFF = 2048;
constexpr int kB = 8, kSsrc = 512, kStgt = 128, kDk = 64;
constexpr int kDD = kD * kD;
constexpr int kMe = kB * kSsrc;   // 4096 encoder tokens
constexpr int kMd = kB * kStgt;   // 1024 decoder tokens
}

// ---------------------------------------------------------------------------
// WMMA helper: D = A(16x32 f16) * B(32x16 f16) + C(16x16 f32)
// ---------------------------------------------------------------------------
__device__ inline v8f wmma32(v16h a, v16h b, v8f c) {
  return __builtin_amdgcn_wmma_f32_16x16x32_f16(false, a, false, b, (short)0, c,
                                                false, false);
}

// A fragment from LDS tile (row-major, padded stride 40 halves).
// lane m=l&15 holds row m; halves 0..7 = K[(l>>4)*8 .. +7], 8..15 = K[16+(l>>4)*8 .. +7]
__device__ inline v16h frag_a(const half_t* S, int mrow, int lane) {
  const half_t* p = S + (mrow + (lane & 15)) * 40;
  int h8 = (lane >> 4) * 8;
  union { v16h v; v8h h[2]; } u;
  u.h[0] = *(const v8h*)(p + h8);
  u.h[1] = *(const v8h*)(p + 16 + h8);
  return u.v;
}
// B fragment: operand is W (N x K row-major) so B = W^T implicitly.
// lane n=l&15 holds W row n; halves 0..7 = K[(l>>4)*8 .. +7], 8..15 = K[16+(l>>4)*8 .. +7]
__device__ inline v16h frag_b(const half_t* S, int nrow, int lane) {
  const half_t* p = S + (nrow + (lane & 15)) * 40 + (lane >> 4) * 8;
  union { v16h v; v8h h[2]; } u;
  u.h[0] = *(const v8h*)(p);
  u.h[1] = *(const v8h*)(p + 16);
  return u.v;
}

#if USE_ASYNC_LDS
// 16B global -> LDS async copy (per-lane addresses, ASYNCcnt-tracked).
__device__ inline void async_cp16(half_t* lds, const half_t* g) {
  unsigned l = (unsigned)(uintptr_t)lds;          // low 32 bits = LDS byte offset
  unsigned long long ga = (unsigned long long)(uintptr_t)g;
  asm volatile("global_load_async_to_lds_b128 %0, %1, off"
               :: "v"(l), "v"(ga) : "memory");
}
__device__ inline void async_wait() {
  asm volatile("s_wait_asynccnt 0x0" ::: "memory");
}
#endif

// ---------------------------------------------------------------------------
// Batched GEMM:  C[z] = act( alpha * A[z] @ B[z]^T + bias )
//   A: M x K (row-major, lda), per-z offset b*aB + h*aH  (z = b*Hn + h)
//   B: N x K (row-major, ldb), per-z offset b*bB + h*bH
//   Cf (fp32, optional), Ch (f16, optional)
// BMxBN block tile, WTMxWTN wave tile, double-buffered LDS, 1 barrier / K-step.
// Requires M % BM == 0, N % BN == 0, K % 32 == 0 (true at all call sites).
// ---------------------------------------------------------------------------
template <int BM, int BN, int WTM, int WTN>
__global__ void __launch_bounds__(2 * BM)
gemm_nt_kernel(const half_t* __restrict__ A, int lda, long aB, long aH,
               const half_t* __restrict__ Bm, int ldb, long bB, long bH,
               float* __restrict__ Cf, int ldc, long cB, long cH,
               half_t* __restrict__ Ch, int ldch, long chB, long chH,
               const float* __restrict__ bias,
               int M, int N, int K, float alpha, int act, int Hn) {
  constexpr int NWN = BN / WTN;                 // waves along N
  constexpr int THREADS = 2 * BM;
  constexpr int MT = WTM / 16, NT = WTN / 16;   // wmma tiles per wave
  static_assert(BM == BN, "square block tile");
  static_assert((BM / WTM) * (BN / WTN) * 32 == THREADS, "wave layout");
  (void)M; (void)N;

  __shared__ __attribute__((aligned(16))) half_t As[2][BM * 40];
  __shared__ __attribute__((aligned(16))) half_t Bs[2][BN * 40];

  const int tid = threadIdx.x, lane = tid & 31, wv = tid >> 5;
  const int wrow = (wv / NWN) * WTM;
  const int wcol = (wv % NWN) * WTN;
  const int zb = blockIdx.z / Hn, zh = blockIdx.z % Hn;
  const long m0 = (long)blockIdx.y * BM;
  const long n0 = (long)blockIdx.x * BN;

  const int fr = tid >> 1;          // 0..BM-1 : tile row this thread fills
  const int fc = (tid & 1) * 16;    // 0/16    : half-chunk within the 32-K row

  const half_t* ga = A  + (size_t)zb * aB + (size_t)zh * aH + (size_t)(m0 + fr) * lda + fc;
  const half_t* gb = Bm + (size_t)zb * bB + (size_t)zh * bH + (size_t)(n0 + fr) * ldb + fc;

  v8f acc[MT][NT];
  const v8f vzero = {0.f, 0.f, 0.f, 0.f, 0.f, 0.f, 0.f, 0.f};
#pragma unroll
  for (int i = 0; i < MT; ++i)
#pragma unroll
    for (int j = 0; j < NT; ++j) acc[i][j] = vzero;

#if USE_ASYNC_LDS
  async_cp16(&As[0][fr * 40 + fc],     ga);
  async_cp16(&As[0][fr * 40 + fc + 8], ga + 8);
  async_cp16(&Bs[0][fr * 40 + fc],     gb);
  async_cp16(&Bs[0][fr * 40 + fc + 8], gb + 8);
#else
  v8h ra0 = *(const v8h*)ga, ra1 = *(const v8h*)(ga + 8);
  v8h rb0 = *(const v8h*)gb, rb1 = *(const v8h*)(gb + 8);
#endif

  for (int kt = 0; kt < K; kt += 32) {
    const int p = (kt >> 5) & 1;
#if USE_ASYNC_LDS
    async_wait();                       // this wave's tile copies done
    __syncthreads();                    // everyone's copies done
    if (kt + 32 < K) {                  // prefetch next tile into other buffer
      ga += 32; gb += 32;
      async_cp16(&As[p ^ 1][fr * 40 + fc],     ga);
      async_cp16(&As[p ^ 1][fr * 40 + fc + 8], ga + 8);
      async_cp16(&Bs[p ^ 1][fr * 40 + fc],     gb);
      async_cp16(&Bs[p ^ 1][fr * 40 + fc + 8], gb + 8);
    }
#else
    *(v8h*)&As[p][fr * 40 + fc]     = ra0;
    *(v8h*)&As[p][fr * 40 + fc + 8] = ra1;
    *(v8h*)&Bs[p][fr * 40 + fc]     = rb0;
    *(v8h*)&Bs[p][fr * 40 + fc + 8] = rb1;
    __syncthreads();
    if (kt + 32 < K) {
      ga += 32; gb += 32;
      ra0 = *(const v8h*)ga; ra1 = *(const v8h*)(ga + 8);
      rb0 = *(const v8h*)gb; rb1 = *(const v8h*)(gb + 8);
    }
#endif
    v16h af[MT], bf[NT];
#pragma unroll
    for (int i = 0; i < MT; ++i) af[i] = frag_a(&As[p][0], wrow + i * 16, lane);
#pragma unroll
    for (int j = 0; j < NT; ++j) bf[j] = frag_b(&Bs[p][0], wcol + j * 16, lane);
#pragma unroll
    for (int i = 0; i < MT; ++i)
#pragma unroll
      for (int j = 0; j < NT; ++j)
        acc[i][j] = wmma32(af[i], bf[j], acc[i][j]);
  }

  float*  cf = Cf ? Cf + (size_t)zb * cB  + (size_t)zh * cH  : nullptr;
  half_t* ch = Ch ? Ch + (size_t)zb * chB + (size_t)zh * chH : nullptr;
  const int lnn = lane & 15, lh = lane >> 4;

#pragma unroll
  for (int i = 0; i < MT; ++i)
#pragma unroll
    for (int j = 0; j < NT; ++j) {
      const long gm0 = m0 + wrow + i * 16 + lh * 8;
      const long gn  = n0 + wcol + j * 16 + lnn;
      const float bv = bias ? bias[gn] : 0.f;
#pragma unroll
      for (int r = 0; r < 8; ++r) {
        const long gm = gm0 + r;
        float v = acc[i][j][r] * alpha + bv;
        if (act == 1) v = 0.5f * v * (1.0f + erff(v * 0.70710678118f));  // exact GELU
        if (cf) cf[gm * (size_t)ldc  + gn] = v;
        if (ch) ch[gm * (size_t)ldch + gn] = (half_t)v;
      }
    }
}

// ---------------------------------------------------------------------------
// Elementwise / reduction kernels
// ---------------------------------------------------------------------------
__global__ void cvt_kernel(const float* __restrict__ s, half_t* __restrict__ d, long n) {
  long i = (long)blockIdx.x * blockDim.x + threadIdx.x;
  long st = (long)gridDim.x * blockDim.x;
  for (; i < n; i += st) d[i] = (half_t)s[i];
}

__global__ void embed_kernel(const int* __restrict__ ids, const float* __restrict__ emb,
                             const float* __restrict__ pos, float* __restrict__ x,
                             half_t* __restrict__ x16, int S, long total) {
  long i = (long)blockIdx.x * blockDim.x + threadIdx.x;
  long st = (long)gridDim.x * blockDim.x;
  for (; i < total; i += st) {
    long row = i / kD; int d = (int)(i % kD); int s2 = (int)(row % S);
    float v = emb[(size_t)ids[row] * kD + d] + pos[(size_t)s2 * kD + d];
    x[i] = v; x16[i] = (half_t)v;
  }
}

// vt[z, d, s] = v[(b*Sk+s)*kD + h*kDk + d],  z = b*kH + h
__global__ void vtrans_kernel(const half_t* __restrict__ v, half_t* __restrict__ vt,
                              int Sk, long total) {
  long i = (long)blockIdx.x * blockDim.x + threadIdx.x;
  long st = (long)gridDim.x * blockDim.x;
  for (; i < total; i += st) {
    long z = i / ((long)kDk * Sk);
    long rem = i % ((long)kDk * Sk);
    long d = rem / Sk, s2 = rem % Sk;
    long b = z / kH, h = z % kH;
    vt[i] = v[(b * Sk + s2) * (long)kD + h * kDk + d];
  }
}

// scores [Z,Sq,Sk] f32 -> attn [Z,Sq,Sk] f16 ; mask: -1e9 then softmax (matches ref)
__global__ void __launch_bounds__(128)
mask_softmax_kernel(const float* __restrict__ scores, half_t* __restrict__ attn,
                    const int* __restrict__ mask, int Sq, int Sk, int causal) {
  const int q = blockIdx.x, z = blockIdx.y, tid = threadIdx.x;
  const int b = z / kH;
  const float* srow = scores + ((size_t)z * Sq + q) * Sk;
  half_t* arow = attn + ((size_t)z * Sq + q) * Sk;
  __shared__ float buf[512];
  __shared__ float red[128];

  float mx = -3.402823e38f;
  for (int k = tid; k < Sk; k += 128) {
    float v = srow[k];
    bool ok = causal ? (k <= q) : (mask[(size_t)b * Sk + k] != 0);
    v = ok ? v : -1.0e9f;
    buf[k] = v;
    mx = fmaxf(mx, v);
  }
  red[tid] = mx; __syncthreads();
  for (int s = 64; s > 0; s >>= 1) { if (tid < s) red[tid] = fmaxf(red[tid], red[tid + s]); __syncthreads(); }
  mx = red[0]; __syncthreads();

  float sum = 0.f;
  for (int k = tid; k < Sk; k += 128) { float e = expf(buf[k] - mx); buf[k] = e; sum += e; }
  red[tid] = sum; __syncthreads();
  for (int s = 64; s > 0; s >>= 1) { if (tid < s) red[tid] += red[tid + s]; __syncthreads(); }
  const float inv = 1.f / red[0];
  for (int k = tid; k < Sk; k += 128) arow[k] = (half_t)(buf[k] * inv);
}

// x = LN(x + a) * gamma + beta ; writes fp32 residual and f16 GEMM operand. D = 512.
__global__ void __launch_bounds__(128)
add_ln_kernel(float* __restrict__ x, const float* __restrict__ a,
              const float* __restrict__ g, const float* __restrict__ bt,
              half_t* __restrict__ x16) {
  const int row = blockIdx.x, tid = threadIdx.x;
  const size_t base = (size_t)row * kD;
  __shared__ float t[kD];
  __shared__ float red[128];

  float s = 0.f, s2 = 0.f;
  for (int d = tid; d < kD; d += 128) {
    float v = x[base + d] + a[base + d];
    t[d] = v; s += v; s2 += v * v;
  }
  red[tid] = s; __syncthreads();
  for (int st = 64; st > 0; st >>= 1) { if (tid < st) red[tid] += red[tid + st]; __syncthreads(); }
  const float mean = red[0] / (float)kD; __syncthreads();
  red[tid] = s2; __syncthreads();
  for (int st = 64; st > 0; st >>= 1) { if (tid < st) red[tid] += red[tid + st]; __syncthreads(); }
  const float var = red[0] / (float)kD - mean * mean;
  const float rstd = rsqrtf(var + 1e-5f);
  for (int d = tid; d < kD; d += 128) {
    float o = g[d] * (t[d] - mean) * rstd + bt[d];
    x[base + d] = o; x16[base + d] = (half_t)o;
  }
}

// ---------------------------------------------------------------------------
// Host orchestration
// ---------------------------------------------------------------------------
static inline size_t align256(size_t x) { return (x + 255) & ~(size_t)255; }

extern "C" void kernel_launch(void* const* d_in, const int* in_sizes, int n_in,
                              void* d_out, int out_size, void* d_ws, size_t ws_size,
                              hipStream_t stream) {
  (void)in_sizes; (void)n_in; (void)out_size; (void)ws_size;

  const int*   input_ids  = (const int*)d_in[0];
  const int*   dec_ids    = (const int*)d_in[1];
  const int*   amask      = (const int*)d_in[2];
  const float* emb        = (const float*)d_in[3];
  const float* pos        = (const float*)d_in[4];
  const float* enc_attn_w = (const float*)d_in[5];
  const float* enc_attn_b = (const float*)d_in[6];
  const float* enc_ln1    = (const float*)d_in[7];
  const float* enc_ff1_w  = (const float*)d_in[8];
  const float* enc_ff1_b  = (const float*)d_in[9];
  const float* enc_ff2_w  = (const float*)d_in[10];
  const float* enc_ff2_b  = (const float*)d_in[11];
  const float* enc_ln2    = (const float*)d_in[12];
  const float* dec_self_w = (const float*)d_in[13];
  const float* dec_self_b = (const float*)d_in[14];
  const float* dec_ln1    = (const float*)d_in[15];
  const float* dec_cross_w= (const float*)d_in[16];
  const float* dec_cross_b= (const float*)d_in[17];
  const float* dec_ln2    = (const float*)d_in[18];
  const float* dec_ff1_w  = (const float*)d_in[19];
  const float* dec_ff1_b  = (const float*)d_in[20];
  const float* dec_ff2_w  = (const float*)d_in[21];
  const float* dec_ff2_b  = (const float*)d_in[22];
  const float* dec_ln3    = (const float*)d_in[23];
  const float* out_w      = (const float*)d_in[24];
  const float* out_b      = (const float*)d_in[25];
  float* logits = (float*)d_out;

  // ---- carve workspace (~285 MB) ----
  char* p = (char*)d_ws;
  auto carveH = [&](size_t n) { half_t* r = (half_t*)p; p += align256(n * sizeof(half_t)); return r; };
  auto carveF = [&](size_t n) { float*  r = (float*) p; p += align256(n * sizeof(float));  return r; };

  half_t* w_enc_attn  = carveH((size_t)kL * 4 * kDD);
  half_t* w_enc_ff1   = carveH((size_t)kL * kDFF * kD);
  half_t* w_enc_ff2   = carveH((size_t)kL * kD * kDFF);
  half_t* w_dec_self  = carveH((size_t)kL * 4 * kDD);
  half_t* w_dec_cross = carveH((size_t)kL * 4 * kDD);
  half_t* w_dec_ff1   = carveH((size_t)kL * kDFF * kD);
  half_t* w_dec_ff2   = carveH((size_t)kL * kD * kDFF);
  half_t* w_out       = carveH((size_t)kV * kD);

  float*  Xe   = carveF((size_t)kMe * kD);
  half_t* Xe16 = carveH((size_t)kMe * kD);
  float*  Yd   = carveF((size_t)kMd * kD);
  half_t* Yd16 = carveH((size_t)kMd * kD);
  half_t* q16  = carveH((size_t)kMe * kD);
  half_t* k16  = carveH((size_t)kMe * kD);
  half_t* v16  = carveH((size_t)kMe * kD);
  half_t* vt16 = carveH((size_t)kB * kH * kDk * kSsrc);
  float*  scores = carveF((size_t)kB * kH * kSsrc * kSsrc);
  half_t* attn16 = carveH((size_t)kB * kH * kSsrc * kSsrc);
  half_t* ctx16  = carveH((size_t)kMe * kD);
  half_t* h16    = carveH((size_t)kMe * kDFF);
  float*  aout   = carveF((size_t)kMe * kD);

  // ---- helpers ----
  auto cvt = [&](const float* s, half_t* d, size_t n) {
    cvt_kernel<<<dim3(4096), dim3(256), 0, stream>>>(s, d, (long)n);
  };
  auto gemm = [&](const half_t* A, int lda, long aB, long aH,
                  const half_t* B, int ldb, long bB, long bH,
                  float* Cf, int ldc, long cB, long cH,
                  half_t* Ch, int ldch, long chB, long chH,
                  const float* bias, int M, int N, int K,
                  float alpha, int act, int Z, int Hn) {
    if ((M % 128) == 0 && (N % 128) == 0) {
      dim3 g(N / 128, M / 128, Z), blk(256);
      gemm_nt_kernel<128, 128, 64, 32><<<g, blk, 0, stream>>>(
          A, lda, aB, aH, B, ldb, bB, bH, Cf, ldc, cB, cH,
          Ch, ldch, chB, chH, bias, M, N, K, alpha, act, Hn);
    } else {
      dim3 g(N / 64, M / 64, Z), blk(128);
      gemm_nt_kernel<64, 64, 32, 32><<<g, blk, 0, stream>>>(
          A, lda, aB, aH, B, ldb, bB, bH, Cf, ldc, cB, cH,
          Ch, ldch, chB, chH, bias, M, N, K, alpha, act, Hn);
    }
  };

  auto mha = [&](const half_t* xq, int Sq, const half_t* xkv, int Skv,
                 const half_t* w4, const float* b4, const int* maskp, int causal) {
    const int Mq = kB * Sq, Mkv = kB * Skv;
    const half_t *wq = w4, *wk = w4 + kDD, *wv = w4 + 2 * kDD, *wo = w4 + 3 * kDD;
    // projections (f16 outputs)
    gemm(xq,  kD,0,0, wq, kD,0,0, nullptr,0,0,0, q16, kD,0,0, b4 + 0*kD, Mq,  kD, kD, 1.f, 0, 1, 1);
    gemm(xkv, kD,0,0, wk, kD,0,0, nullptr,0,0,0, k16, kD,0,0, b4 + 1*kD, Mkv, kD, kD, 1.f, 0, 1, 1);
    gemm(xkv, kD,0,0, wv, kD,0,0, nullptr,0,0,0, v16, kD,0,0, b4 + 2*kD, Mkv, kD, kD, 1.f, 0, 1, 1);
    vtrans_kernel<<<dim3(2048), dim3(256), 0, stream>>>(v16, vt16, Skv, (long)kB * kH * kDk * Skv);
    // scores = (Q @ K^T) * sqrt(dk)   [reference multiplies by sqrt(dk)]
    gemm(q16, kD, (long)Sq  * kD, kDk,
         k16, kD, (long)Skv * kD, kDk,
         scores, Skv, (long)kH * Sq * Skv, (long)Sq * Skv,
         nullptr, 0, 0, 0,
         nullptr, Sq, Skv, kDk, 8.0f, 0, kB * kH, kH);
    mask_softmax_kernel<<<dim3(Sq, kB * kH), dim3(128), 0, stream>>>(scores, attn16, maskp, Sq, Skv, causal);
    // ctx = attn @ V   (B operand is V^T, rows = head dims) ; N=64 -> 64x64 path
    gemm(attn16, Skv, (long)kH * Sq * Skv, (long)Sq * Skv,
         vt16,   Skv, (long)kH * kDk * Skv, (long)kDk * Skv,
         nullptr, 0, 0, 0,
         ctx16, kD, (long)Sq * kD, kDk,
         nullptr, Sq, kDk, Skv, 1.f, 0, kB * kH, kH);
    // output projection -> aout (fp32)
    gemm(ctx16, kD,0,0, wo, kD,0,0, aout, kD,0,0, nullptr,0,0,0, b4 + 3*kD, Mq, kD, kD, 1.f, 0, 1, 1);
  };

  auto ffn = [&](const half_t* x, int M, const half_t* w1, const float* b1,
                 const half_t* w2, const float* b2) {
    gemm(x,   kD,  0,0, w1, kD,  0,0, nullptr,0,0,0, h16, kDFF,0,0, b1, M, kDFF, kD,  1.f, 1, 1, 1);
    gemm(h16, kDFF,0,0, w2, kDFF,0,0, aout, kD,0,0,  nullptr,0,0,0, b2, M, kD,  kDFF, 1.f, 0, 1, 1);
  };

  auto ln = [&](float* x, half_t* x16p, int M, const float* lnp) {
    add_ln_kernel<<<dim3(M), dim3(128), 0, stream>>>(x, aout, lnp, lnp + kD, x16p);
  };

  // ---- weight conversion to f16 ----
  cvt(enc_attn_w,  w_enc_attn,  (size_t)kL * 4 * kDD);
  cvt(enc_ff1_w,   w_enc_ff1,   (size_t)kL * kDFF * kD);
  cvt(enc_ff2_w,   w_enc_ff2,   (size_t)kL * kD * kDFF);
  cvt(dec_self_w,  w_dec_self,  (size_t)kL * 4 * kDD);
  cvt(dec_cross_w, w_dec_cross, (size_t)kL * 4 * kDD);
  cvt(dec_ff1_w,   w_dec_ff1,   (size_t)kL * kDFF * kD);
  cvt(dec_ff2_w,   w_dec_ff2,   (size_t)kL * kD * kDFF);
  cvt(out_w,       w_out,       (size_t)kV * kD);

  // ---- encoder ----
  embed_kernel<<<dim3(4096), dim3(256), 0, stream>>>(input_ids, emb, pos, Xe, Xe16, kSsrc, (long)kMe * kD);
  for (int i = 0; i < kL; ++i) {
    mha(Xe16, kSsrc, Xe16, kSsrc, w_enc_attn + (size_t)i * 4 * kDD,
        enc_attn_b + (size_t)i * 4 * kD, amask, 0);
    ln(Xe, Xe16, kMe, enc_ln1 + (size_t)i * 2 * kD);
    ffn(Xe16, kMe, w_enc_ff1 + (size_t)i * kDFF * kD, enc_ff1_b + (size_t)i * kDFF,
        w_enc_ff2 + (size_t)i * kD * kDFF, enc_ff2_b + (size_t)i * kD);
    ln(Xe, Xe16, kMe, enc_ln2 + (size_t)i * 2 * kD);
  }
  // Xe16 now holds enc_out (f16), used by cross-attention.

  // ---- decoder ----
  embed_kernel<<<dim3(1024), dim3(256), 0, stream>>>(dec_ids, emb, pos, Yd, Yd16, kStgt, (long)kMd * kD);
  for (int i = 0; i < kL; ++i) {
    mha(Yd16, kStgt, Yd16, kStgt, w_dec_self + (size_t)i * 4 * kDD,
        dec_self_b + (size_t)i * 4 * kD, nullptr, 1);
    ln(Yd, Yd16, kMd, dec_ln1 + (size_t)i * 2 * kD);
    mha(Yd16, kStgt, Xe16, kSsrc, w_dec_cross + (size_t)i * 4 * kDD,
        dec_cross_b + (size_t)i * 4 * kD, amask, 0);
    ln(Yd, Yd16, kMd, dec_ln2 + (size_t)i * 2 * kD);
    ffn(Yd16, kMd, w_dec_ff1 + (size_t)i * kDFF * kD, dec_ff1_b + (size_t)i * kDFF,
        w_dec_ff2 + (size_t)i * kD * kDFF, dec_ff2_b + (size_t)i * kD);
    ln(Yd, Yd16, kMd, dec_ln3 + (size_t)i * 2 * kD);
  }

  // ---- vocab projection: logits = Y @ out_w^T + out_b (fp32 out) ----
  gemm(Yd16, kD, 0, 0, w_out, kD, 0, 0,
       logits, kV, 0, 0, nullptr, 0, 0, 0,
       out_b, kMd, kV, kD, 1.f, 0, 1, 1);
}
